// MAICMultiHeadAgent_88648124989768
// MI455X (gfx1250) — compile-verified, matrix operand order checked by reading, not run on hardware
//
#include <hip/hip_runtime.h>
#include <hip/hip_bf16.h>

// ---------------------------------------------------------------------------
// Problem constants (from the reference)
// ---------------------------------------------------------------------------
#define BSZ   1024
#define NN    16
#define INF   256
#define HF    256
#define NHF   256
#define LF    64
#define AF    32
#define HEADS 4
#define ADIM  128
#define HD    32
#define ROWS  (BSZ * NN)          // 16384
#define ROWS2 (BSZ * NN * NN)     // 262144

typedef __attribute__((ext_vector_type(16))) __bf16 v16bf;
typedef __attribute__((ext_vector_type(8)))  __bf16 v8bf;
typedef __attribute__((ext_vector_type(4)))  __bf16 v4bf;
typedef __attribute__((ext_vector_type(8)))  float  v8f;

__device__ __forceinline__ v8bf pack8(float4 a, float4 b) {
    v8bf r;
    r[0] = (__bf16)a.x; r[1] = (__bf16)a.y; r[2] = (__bf16)a.z; r[3] = (__bf16)a.w;
    r[4] = (__bf16)b.x; r[5] = (__bf16)b.y; r[6] = (__bf16)b.z; r[7] = (__bf16)b.w;
    return r;
}
__device__ __forceinline__ v4bf pack4(float4 a) {
    v4bf r;
    r[0] = (__bf16)a.x; r[1] = (__bf16)a.y; r[2] = (__bf16)a.z; r[3] = (__bf16)a.w;
    return r;
}
__device__ __forceinline__ v16bf cat16(v8bf lo, v8bf hi) {
    return __builtin_shufflevector(lo, hi, 0, 1, 2, 3, 4, 5, 6, 7,
                                           8, 9, 10, 11, 12, 13, 14, 15);
}

// ---------------------------------------------------------------------------
// WMMA GEMM:  C[M,N] = act( concat(A1[row>>4], A2[row]) @ W[N,K]^T + bias )
// Block = 256 threads = 8 wave32 waves: 4 along M (32 rows each, 2 acc tiles)
// x 2 along N (16 cols each).  Block tile = 128 x 32, K step 32.
// Requires M%128==0, N%32==0, K1%32==0, K2%32==0.
// ---------------------------------------------------------------------------
#define TM 128
#define TN 32
#define TK 32
#define LDP 40            // LDS row stride in bf16 (80 B, keeps 16 B alignment)

template<int ACT, bool CONCAT>   // ACT: 0 none, 1 relu, 2 leaky(0.01)
__global__ __launch_bounds__(256)
void gemm_bias_act(const float* __restrict__ A1, const float* __restrict__ A2,
                   const float* __restrict__ W,  const float* __restrict__ bias,
                   float* __restrict__ C,
                   int M, int N, int K1, int K2)
{
    __shared__ __bf16 As[TM * LDP];
    __shared__ __bf16 Bs[TN * LDP];

    const int K    = CONCAT ? (K1 + K2) : K1;
    const int tid  = threadIdx.x;
    const int lane = tid & 31;
    const int wave = tid >> 5;
    const int wm   = wave & 3;          // 4 waves along M (32 rows each)
    const int wn   = wave >> 2;         // 2 waves along N (16 cols each)
    const int rowbase = blockIdx.y * TM;
    const int colbase = blockIdx.x * TN;

    const int half = lane >> 4;         // 0 / 1
    const int l16  = lane & 15;
    const int kb   = half * 8;          // K sub-offset per ISA 16-bit layout

    // A loader: thread covers 16 contiguous K of one row
    const int ar = tid >> 1;
    const int ak = (tid & 1) * 16;
    // B loader: thread covers 4 contiguous K of one row of W
    const int bn = tid >> 3;
    const int bk = (tid & 7) * 4;

    v8f acc0 = {}, acc1 = {};

    for (int k0 = 0; k0 < K; k0 += TK) {
        // ---- A tile: 4 x b128 load, pack, 2 x b128 LDS store
        {
            const float* src;
            int ldk, kk0;
            bool shr;
            if (!CONCAT || k0 < K1) { src = A1; ldk = K1; kk0 = k0;      shr = CONCAT; }
            else                    { src = A2; ldk = K2; kk0 = k0 - K1; shr = false;  }
            int grow = rowbase + ar;
            if (shr) grow >>= 4;        // rowdiv = 16 (h broadcast over j)
            const float4* p = (const float4*)(src + (size_t)grow * ldk + kk0 + ak);
            float4 f0 = p[0], f1 = p[1], f2 = p[2], f3 = p[3];
            *(v8bf*)&As[ar * LDP + ak]     = pack8(f0, f1);
            *(v8bf*)&As[ar * LDP + ak + 8] = pack8(f2, f3);
        }
        // ---- B tile: 1 x b128 load, pack, b64 LDS store
        {
            const float4* p = (const float4*)(W + (size_t)(colbase + bn) * K + k0 + bk);
            *(v4bf*)&Bs[bn * LDP + bk] = pack4(p[0]);
        }
        __syncthreads();

        // ---- fragments (2 x ds_load_b128 each), per ISA 7.12.2 16-bit layout
        const int a0r = (wm * 32 + l16) * LDP;
        const int a1r = a0r + 16 * LDP;
        const int bbr = (wn * 16 + l16) * LDP;
        v16bf a0 = cat16(*(v8bf*)&As[a0r + kb], *(v8bf*)&As[a0r + kb + 16]);
        v16bf a1 = cat16(*(v8bf*)&As[a1r + kb], *(v8bf*)&As[a1r + kb + 16]);
        v16bf b  = cat16(*(v8bf*)&Bs[bbr + kb], *(v8bf*)&Bs[bbr + kb + 16]);

        acc0 = __builtin_amdgcn_wmma_f32_16x16x32_bf16(
                   false, a0, false, b, (short)0, acc0, false, false);
        acc1 = __builtin_amdgcn_wmma_f32_16x16x32_bf16(
                   false, a1, false, b, (short)0, acc1, false, false);
        __syncthreads();
    }

    // ---- epilogue: bias + activation; C layout (VGPR r -> row r / r+8)
    const int col = colbase + wn * 16 + l16;
    const float bv = bias[col];
    #pragma unroll
    for (int r = 0; r < 8; ++r) {
        int row0 = rowbase + wm * 32 + r + half * 8;
        float v0 = acc0[r] + bv;
        float v1 = acc1[r] + bv;
        if (ACT == 1) { v0 = v0 > 0.0f ? v0 : 0.0f;        v1 = v1 > 0.0f ? v1 : 0.0f; }
        if (ACT == 2) { v0 = v0 > 0.0f ? v0 : 0.01f * v0;  v1 = v1 > 0.0f ? v1 : 0.01f * v1; }
        C[(size_t)row0 * N + col]        = v0;
        C[(size_t)(row0 + 16) * N + col] = v1;
    }
}

// ---------------------------------------------------------------------------
// GRU gate fusion:  h = (1-z)*n + z*hs     (PyTorch gate order r,z,n)
// ---------------------------------------------------------------------------
__global__ void gru_kernel(const float* __restrict__ gx, const float* __restrict__ gh,
                           const float* __restrict__ hs, float* __restrict__ h, int total)
{
    int idx = blockIdx.x * blockDim.x + threadIdx.x;
    if (idx >= total) return;
    int row = idx >> 8;
    int c   = idx & 255;
    const float* gxr = gx + (size_t)row * (3 * HF);
    const float* ghr = gh + (size_t)row * (3 * HF);
    float r = 1.0f / (1.0f + __expf(-(gxr[c]      + ghr[c])));
    float z = 1.0f / (1.0f + __expf(-(gxr[HF + c] + ghr[HF + c])));
    float n = tanhf(gxr[2 * HF + c] + r * ghr[2 * HF + c]);
    h[idx] = (1.0f - z) * n + z * hs[idx];
}

// ---------------------------------------------------------------------------
// Training-mode BatchNorm1d over rows + LeakyReLU, in place. Block per column.
// ---------------------------------------------------------------------------
__global__ __launch_bounds__(256)
void bn_leaky_kernel(float* __restrict__ t, const float* __restrict__ g,
                     const float* __restrict__ b, int rows, int cols)
{
    __shared__ float ssum[256], ssq[256];
    const int c = blockIdx.x;
    float s = 0.0f, sq = 0.0f;
    for (int r = threadIdx.x; r < rows; r += 256) {
        float v = t[(size_t)r * cols + c];
        s += v; sq += v * v;
    }
    ssum[threadIdx.x] = s; ssq[threadIdx.x] = sq;
    __syncthreads();
    for (int off = 128; off > 0; off >>= 1) {
        if (threadIdx.x < off) {
            ssum[threadIdx.x] += ssum[threadIdx.x + off];
            ssq[threadIdx.x]  += ssq[threadIdx.x + off];
        }
        __syncthreads();
    }
    const float mean  = ssum[0] / (float)rows;
    const float var   = ssq[0] / (float)rows - mean * mean;
    const float scale = g[c] * rsqrtf(var + 1e-5f);
    const float shift = b[c];
    for (int r = threadIdx.x; r < rows; r += 256) {
        size_t i = (size_t)r * cols + c;
        float v = (t[i] - mean) * scale + shift;
        t[i] = v > 0.0f ? v : 0.01f * v;
    }
}

// ---------------------------------------------------------------------------
// Latent rsample:  latent = lmean + sqrt(max(exp(lvar), 0.002)) * noise
// ---------------------------------------------------------------------------
__global__ void latent_kernel(const float* __restrict__ lp, const float* __restrict__ noise,
                              float* __restrict__ latent, int total)
{
    int idx = blockIdx.x * blockDim.x + threadIdx.x;
    if (idx >= total) return;
    int l  = idx & (LF - 1);
    int r2 = idx >> 6;
    int j  = r2 & (NN - 1);
    int bi = r2 >> 4;
    const float* row = lp + (size_t)bi * (2 * NN * LF);
    float mean = row[j * LF + l];
    float lv   = row[NN * LF + j * LF + l];
    float var  = fmaxf(__expf(lv), 0.002f);
    latent[idx] = mean + sqrtf(var) * noise[idx];
}

// ---------------------------------------------------------------------------
// Masked softmax over j:  alpha[b,i,j,h].  Thread per (b,i,h).
// ---------------------------------------------------------------------------
__global__ void alpha_kernel(const float* __restrict__ query, const float* __restrict__ key,
                             float* __restrict__ alpha, int total)
{
    int idx = blockIdx.x * blockDim.x + threadIdx.x;
    if (idx >= total) return;
    int hh = idx & (HEADS - 1);
    int bi = idx >> 2;
    int i  = bi & (NN - 1);
    const float* qv = query + (size_t)bi * ADIM + hh * HD;
    float sc[NN];
    float mx = -1e30f;
    #pragma unroll
    for (int j = 0; j < NN; ++j) {
        const float* kv = key + ((size_t)bi * NN + j) * ADIM + hh * HD;
        float d = 0.0f;
        #pragma unroll
        for (int dd = 0; dd < HD; ++dd) d += qv[dd] * kv[dd];
        d *= 0.17677669529663687f;          // 1/sqrt(32)
        if (j == i) d = -1e9f;
        sc[j] = d;
        mx = fmaxf(mx, d);
    }
    float sum = 0.0f;
    #pragma unroll
    for (int j = 0; j < NN; ++j) { sc[j] = __expf(sc[j] - mx); sum += sc[j]; }
    const float inv = 1.0f / sum;
    #pragma unroll
    for (int j = 0; j < NN; ++j)
        alpha[((size_t)bi * NN + j) * HEADS + hh] = sc[j] * inv;
}

// ---------------------------------------------------------------------------
// head_messages[b,j,h,a] = sum_i alpha[b,i,j,h] * msg[b,i,j,h,a]
// ---------------------------------------------------------------------------
__global__ void hm_kernel(const float* __restrict__ alpha, const float* __restrict__ msg,
                          float* __restrict__ hm, int total)
{
    int idx = blockIdx.x * blockDim.x + threadIdx.x;
    if (idx >= total) return;
    int ca = idx & (ADIM - 1);
    int bj = idx >> 7;
    int j  = bj & (NN - 1);
    int b  = bj >> 4;
    int hh = ca >> 5;
    float s = 0.0f;
    #pragma unroll
    for (int i = 0; i < NN; ++i) {
        size_t r2 = (size_t)(b * NN + i) * NN + j;
        s += alpha[r2 * HEADS + hh] * msg[r2 * ADIM + ca];
    }
    hm[idx] = s;
}

// ---------------------------------------------------------------------------
// Final:  out[0:524288] = q + comm ;  out[524288:] = h
// ---------------------------------------------------------------------------
__global__ void final_kernel(const float* __restrict__ q, const float* __restrict__ comm,
                             const float* __restrict__ h, float* __restrict__ out,
                             int qn, int total)
{
    int idx = blockIdx.x * blockDim.x + threadIdx.x;
    if (idx >= total) return;
    if (idx < qn) out[idx] = q[idx] + comm[idx];
    else          out[idx] = h[idx - qn];
}

// ---------------------------------------------------------------------------
// Host-side orchestration
// ---------------------------------------------------------------------------
extern "C" void kernel_launch(void* const* d_in, const int* in_sizes, int n_in,
                              void* d_out, int out_size, void* d_ws, size_t ws_size,
                              hipStream_t stream)
{
    const float* inputs = (const float*)d_in[0];
    const float* hidden = (const float*)d_in[1];
    const float* noise  = (const float*)d_in[2];
    const float* fc1_w  = (const float*)d_in[3];
    const float* fc1_b  = (const float*)d_in[4];
    const float* w_ih   = (const float*)d_in[5];
    const float* b_ih   = (const float*)d_in[6];
    const float* w_hh   = (const float*)d_in[7];
    const float* b_hh   = (const float*)d_in[8];
    const float* fc2_w  = (const float*)d_in[9];
    const float* fc2_b  = (const float*)d_in[10];
    const float* e1_w   = (const float*)d_in[11];
    const float* e1_b   = (const float*)d_in[12];
    const float* bn_g   = (const float*)d_in[13];
    const float* bn_b   = (const float*)d_in[14];
    const float* e2_w   = (const float*)d_in[15];
    const float* e2_b   = (const float*)d_in[16];
    const float* m1_w   = (const float*)d_in[17];
    const float* m1_b   = (const float*)d_in[18];
    const float* m2_w   = (const float*)d_in[19];
    const float* m2_b   = (const float*)d_in[20];
    const float* wq_w   = (const float*)d_in[21];
    const float* wq_b   = (const float*)d_in[22];
    const float* wk_w   = (const float*)d_in[23];
    const float* wk_b   = (const float*)d_in[24];
    const float* fus_w  = (const float*)d_in[25];
    const float* fus_b  = (const float*)d_in[26];
    float* out = (float*)d_out;
    float* ws  = (float*)d_ws;

    // -------- workspace layout (floats); one big scratch region is reused
    size_t off = 0;
    float* h      = ws + off; off += (size_t)ROWS  * HF;
    float* latent = ws + off; off += (size_t)ROWS2 * LF;
    float* q      = ws + off; off += (size_t)ROWS  * AF;
    float* query  = ws + off; off += (size_t)ROWS  * ADIM;
    float* key    = ws + off; off += (size_t)ROWS2 * ADIM;
    float* msg    = ws + off; off += (size_t)ROWS2 * ADIM;
    float* alpha  = ws + off; off += (size_t)ROWS2 * HEADS;
    float* hm     = ws + off; off += (size_t)ROWS  * ADIM;
    float* comm   = ws + off; off += (size_t)ROWS  * AF;
    float* scr    = ws + off;
    // phase 1: x | gx | gh    phase 2: t | lp    phase 3: m1out (all of scr)
    float* x     = scr;
    float* gx    = scr + (size_t)ROWS * HF;
    float* gh    = gx  + (size_t)ROWS * 3 * HF;
    float* t     = scr;
    float* lp    = scr + (size_t)ROWS * HF;
    float* m1out = scr;

    // 1) x = relu(inputs @ fc1^T + b)
    gemm_bias_act<1, false><<<dim3(HF / TN, ROWS / TM), 256, 0, stream>>>(
        inputs, nullptr, fc1_w, fc1_b, x, ROWS, HF, INF, 0);
    // 2-3) GRU pre-gates
    gemm_bias_act<0, false><<<dim3(3 * HF / TN, ROWS / TM), 256, 0, stream>>>(
        x, nullptr, w_ih, b_ih, gx, ROWS, 3 * HF, HF, 0);
    gemm_bias_act<0, false><<<dim3(3 * HF / TN, ROWS / TM), 256, 0, stream>>>(
        hidden, nullptr, w_hh, b_hh, gh, ROWS, 3 * HF, HF, 0);
    // 4) GRU gate fusion -> h
    gru_kernel<<<(ROWS * HF) / 256, 256, 0, stream>>>(gx, gh, hidden, h, ROWS * HF);
    // 5) q = h @ fc2^T + b
    gemm_bias_act<0, false><<<dim3(AF / TN, ROWS / TM), 256, 0, stream>>>(
        h, nullptr, fc2_w, fc2_b, q, ROWS, AF, HF, 0);
    // 6) t = h @ e1^T + b
    gemm_bias_act<0, false><<<dim3(NHF / TN, ROWS / TM), 256, 0, stream>>>(
        h, nullptr, e1_w, e1_b, t, ROWS, NHF, HF, 0);
    // 7) BatchNorm (training stats) + LeakyReLU in place
    bn_leaky_kernel<<<NHF, 256, 0, stream>>>(t, bn_g, bn_b, ROWS, NHF);
    // 8) lp = t @ e2^T + b
    gemm_bias_act<0, false><<<dim3((2 * NN * LF) / TN, ROWS / TM), 256, 0, stream>>>(
        t, nullptr, e2_w, e2_b, lp, ROWS, 2 * NN * LF, NHF, 0);
    // 9) latent rsample
    latent_kernel<<<(ROWS2 * LF) / 256, 256, 0, stream>>>(lp, noise, latent, ROWS2 * LF);
    // 10) m1out = leaky( concat(h[row>>4], latent[row]) @ m1^T + b )
    gemm_bias_act<2, true><<<dim3(NHF / TN, ROWS2 / TM), 256, 0, stream>>>(
        h, latent, m1_w, m1_b, m1out, ROWS2, NHF, HF, LF);
    // 11) msg = m1out @ m2^T + b
    gemm_bias_act<0, false><<<dim3((HEADS * AF) / TN, ROWS2 / TM), 256, 0, stream>>>(
        m1out, nullptr, m2_w, m2_b, msg, ROWS2, HEADS * AF, NHF, 0);
    // 12) query = h @ wq^T + b
    gemm_bias_act<0, false><<<dim3(ADIM / TN, ROWS / TM), 256, 0, stream>>>(
        h, nullptr, wq_w, wq_b, query, ROWS, ADIM, HF, 0);
    // 13) key = latent @ wk^T + b
    gemm_bias_act<0, false><<<dim3(ADIM / TN, ROWS2 / TM), 256, 0, stream>>>(
        latent, nullptr, wk_w, wk_b, key, ROWS2, ADIM, LF, 0);
    // 14) masked softmax
    alpha_kernel<<<(ROWS * HEADS) / 256, 256, 0, stream>>>(query, key, alpha, ROWS * HEADS);
    // 15) head messages
    hm_kernel<<<(ROWS * ADIM) / 256, 256, 0, stream>>>(alpha, msg, hm, ROWS * ADIM);
    // 16) comm = hm @ fus^T + b
    gemm_bias_act<0, false><<<dim3(AF / TN, ROWS / TM), 256, 0, stream>>>(
        hm, nullptr, fus_w, fus_b, comm, ROWS, AF, ADIM, 0);
    // 17) final concat output
    const int qn = ROWS * AF;
    const int total = qn + ROWS * HF;
    final_kernel<<<(total + 255) / 256, 256, 0, stream>>>(q, comm, h, out, qn, total);
}
